// CrossAttention_8796093022590
// MI455X (gfx1250) — matrix operand order repaired and checked by reading, not compiled, not run on hardware
//
#include <hip/hip_runtime.h>

// ---------------------------------------------------------------------------
// CrossAttention for MI455X (gfx1250, wave32, WMMA bf16 -> f32 accumulate)
// B=4, T=2048, S=512, D=1024, H=16, Hd=64
//
// Pipeline:
//   1. x, context -> bf16 (elementwise)
//   2. Wq, Wkv, Wp -> transposed bf16  [N,K]
//   3. bf16 GEMMs (Q, KV) with 2x2 WMMA register blocking
//   4. flash attention (bf16 WMMA, online softmax)
//   5. bf16 GEMM -> fp32 output projection
// ---------------------------------------------------------------------------

typedef __attribute__((ext_vector_type(16))) __bf16 v16bf;
typedef __attribute__((ext_vector_type(8)))  __bf16 v8bf;
typedef __attribute__((ext_vector_type(8)))  float  v8f;
typedef __attribute__((ext_vector_type(4)))  float  v4f;

__device__ __forceinline__ v8f wmma_bf16(v16bf a, v16bf b, v8f c) {
  return __builtin_amdgcn_wmma_f32_16x16x32_bf16(
      false, a, false, b, (short)0, c, false, false);
}

// A-fragment (16x32 bf16): lane row = M0+l16; elem e -> k0+hf*16+e.
// Two contiguous 16B loads.
__device__ __forceinline__ v16bf load_afrag(const __bf16* p) {
  v8bf lo = *(const v8bf*)(p);
  v8bf hi = *(const v8bf*)(p + 8);
  v16bf r;
#pragma unroll
  for (int e = 0; e < 8; ++e) { r[e] = lo[e]; r[e + 8] = hi[e]; }
  return r;
}

// ---------------------------------------------------------------------------
// elementwise f32 -> bf16 (8 elems / thread)
// ---------------------------------------------------------------------------
__global__ __launch_bounds__(256)
void cvt_f32_bf16(const float* __restrict__ in, __bf16* __restrict__ out,
                  size_t n) {
  size_t i = ((size_t)blockIdx.x * blockDim.x + threadIdx.x) * 8;
  if (i + 8 <= n) {
    v4f a = *(const v4f*)(in + i);
    v4f b = *(const v4f*)(in + i + 4);
    v8bf o;
#pragma unroll
    for (int e = 0; e < 4; ++e) { o[e] = (__bf16)a[e]; o[e + 4] = (__bf16)b[e]; }
    *(v8bf*)(out + i) = o;
  }
}

// ---------------------------------------------------------------------------
// transpose f32 [R,C] -> bf16 [C,R]  (32x32 LDS tiles, block 32x8)
// ---------------------------------------------------------------------------
__global__ __launch_bounds__(256)
void transpose_f32_bf16(const float* __restrict__ in, __bf16* __restrict__ out,
                        int R, int C) {
  __shared__ float tile[32][33];
  const int bx = blockIdx.x * 32;   // col base
  const int by = blockIdx.y * 32;   // row base
  const int tx = threadIdx.x, ty = threadIdx.y;
#pragma unroll
  for (int j = 0; j < 4; ++j)
    tile[ty + j * 8][tx] = in[(size_t)(by + ty + j * 8) * C + bx + tx];
  __syncthreads();
#pragma unroll
  for (int j = 0; j < 4; ++j)
    out[(size_t)(bx + ty + j * 8) * R + by + tx] =
        (__bf16)tile[tx][ty + j * 8];
}

// ---------------------------------------------------------------------------
// GEMM: A bf16 [M,K] row-major, BT bf16 [N,K] row-major (pre-transposed),
// bias f32 [N] -> C [M,N] (OutT = __bf16 or float).
// 8 waves: 2 (m) x 4 (n); each wave computes 2x2 16x16 tiles (32x32).
// Block tile: 64 x 128.
// ---------------------------------------------------------------------------
template <typename OutT>
__global__ __launch_bounds__(256)
void gemm_bf16_t(const __bf16* __restrict__ A, const __bf16* __restrict__ BT,
                 const float* __restrict__ bias, OutT* __restrict__ C,
                 int M, int N, int K) {
  const int wave = threadIdx.x >> 5;
  const int lane = threadIdx.x & 31;
  const int hf   = lane >> 4;
  const int l16  = lane & 15;
  const int m0 = blockIdx.x * 64  + (wave >> 2) * 32;
  const int n0 = blockIdx.y * 128 + (wave & 3) * 32;

  v8f acc[2][2];
#pragma unroll
  for (int i = 0; i < 2; ++i)
#pragma unroll
    for (int j = 0; j < 2; ++j)
      acc[i][j] = (v8f){0.f,0.f,0.f,0.f,0.f,0.f,0.f,0.f};

  const __bf16* aB[2] = {
      A + (size_t)(m0 + l16) * K + hf * 16,
      A + (size_t)(m0 + 16 + l16) * K + hf * 16};
  // B fragment: lane col = n; elem e -> K = k0 + hf*16 + e: 16 contiguous bf16
  const __bf16* bB[2] = {
      BT + (size_t)(n0 + l16) * K + hf * 16,
      BT + (size_t)(n0 + 16 + l16) * K + hf * 16};

  for (int k0 = 0; k0 < K; k0 += 32) {
    __builtin_prefetch(aB[0] + k0 + 128, 0, 1);
    __builtin_prefetch(bB[0] + k0 + 128, 0, 1);
    v16bf a0 = load_afrag(aB[0] + k0);
    v16bf a1 = load_afrag(aB[1] + k0);
    v16bf b0 = *(const v16bf*)(bB[0] + k0);
    v16bf b1 = *(const v16bf*)(bB[1] + k0);
    acc[0][0] = wmma_bf16(a0, b0, acc[0][0]);
    acc[0][1] = wmma_bf16(a0, b1, acc[0][1]);
    acc[1][0] = wmma_bf16(a1, b0, acc[1][0]);
    acc[1][1] = wmma_bf16(a1, b1, acc[1][1]);
  }

#pragma unroll
  for (int j = 0; j < 2; ++j) {
    const int col = n0 + j * 16 + l16;
    const float bv = bias[col];
#pragma unroll
    for (int i = 0; i < 2; ++i) {
#pragma unroll
      for (int r = 0; r < 8; ++r) {
        const int row = m0 + i * 16 + hf * 8 + r;
        C[(size_t)row * N + col] = (OutT)(acc[i][j][r] + bv);
      }
    }
  }
}

// ---------------------------------------------------------------------------
// Flash attention. qbuf: bf16 [B*T, D] (row = b*T+t, col = h*Hd+hd)
// kvbuf: bf16 [B*S, 2D]  (K at cols [0,D), V at cols [D,2D))
// obuf : bf16 [B*T, D]   ("bhts,bhsd->bthd" layout)
// Block: 256 thr (8 waves) -> 128 t-rows of one (b,h). Wave owns 16 t-rows.
// ---------------------------------------------------------------------------
#define HD 64
__global__ __launch_bounds__(256)
void attn_kernel(const __bf16* __restrict__ qbuf, const __bf16* __restrict__ kvbuf,
                 const unsigned char* __restrict__ cmask, __bf16* __restrict__ obuf,
                 int Bn, int T, int S, int H) {
  const int Dm = H * HD;        // 1024
  const int KVs = 2 * Dm;       // kv row stride

  const int wave = threadIdx.x >> 5;
  const int lane = threadIdx.x & 31;
  const int hf   = lane >> 4;
  const int l16  = lane & 15;

  const int bh = blockIdx.y;
  const int b  = bh / H;
  const int h  = bh % H;
  const int t0 = blockIdx.x * 128 + wave * 16;

  __shared__ __bf16 ldsV[HD * 32];      // transposed V tile: [hd][s]
  __shared__ __bf16 ldsP[8][16 * 32];   // per-wave P tile:   [row][s]

  v8f oacc[4];
#pragma unroll
  for (int nt = 0; nt < 4; ++nt)
    oacc[nt] = (v8f){0.f,0.f,0.f,0.f,0.f,0.f,0.f,0.f};

  float rmax[8], rsum[8], corr[8];
#pragma unroll
  for (int r = 0; r < 8; ++r) { rmax[r] = -3.402823466e38f; rsum[r] = 0.f; }

  const float scale = 0.125f;               // 1/sqrt(64)
  const float NEG   = -3.402823466e38f;     // finfo(f32).min

  for (int s0 = 0; s0 < S; s0 += 32) {
    __syncthreads();   // previous iter's LDS reads done before overwrite

    // ---- cooperative V tile load (32 s x 64 hd), transposed into LDS ----
    {
      const int sl  = threadIdx.x >> 3;         // 0..31
      const int hd0 = (threadIdx.x & 7) * 8;    // 0..56
      const __bf16* vp =
          kvbuf + (size_t)(b * S + s0 + sl) * KVs + Dm + h * HD + hd0;
      v8bf vv = *(const v8bf*)vp;
#pragma unroll
      for (int j = 0; j < 8; ++j) ldsV[(hd0 + j) * 32 + sl] = vv[j];
    }

    // ---- scores: two 16x16 tiles covering s0..s0+31 ----
    v8f c0 = {0.f,0.f,0.f,0.f,0.f,0.f,0.f,0.f};
    v8f c1 = {0.f,0.f,0.f,0.f,0.f,0.f,0.f,0.f};
#pragma unroll
    for (int kk = 0; kk < HD; kk += 32) {
      v16bf a = load_afrag(
          qbuf + (size_t)(b * T + t0 + l16) * Dm + h * HD + kk + hf * 16);
#pragma unroll
      for (int nt = 0; nt < 2; ++nt) {
        const int s = s0 + nt * 16 + l16;
        const __bf16* kp =
            kvbuf + (size_t)(b * S + s) * KVs + h * HD + kk + hf * 16;
        v8bf k0v = *(const v8bf*)(kp);
        v8bf k1v = *(const v8bf*)(kp + 8);
        v16bf bf;
#pragma unroll
        for (int e = 0; e < 8; ++e) { bf[e] = k0v[e]; bf[e + 8] = k1v[e]; }
        if (nt == 0) c0 = wmma_bf16(a, bf, c0);
        else         c1 = wmma_bf16(a, bf, c1);
      }
    }

    // ---- scale + context mask (mask is per-column == per-lane) ----
    const bool mk0 = cmask[b * S + s0 + l16] != 0;
    const bool mk1 = cmask[b * S + s0 + 16 + l16] != 0;
#pragma unroll
    for (int r = 0; r < 8; ++r) {
      c0[r] = mk0 ? c0[r] * scale : NEG;
      c1[r] = mk1 ? c1[r] * scale : NEG;
    }

    // ---- online softmax (row = hf*8 + r lives in this lane's 16-lane half) --
#pragma unroll
    for (int r = 0; r < 8; ++r) {
      float mx = fmaxf(c0[r], c1[r]);
#pragma unroll
      for (int off = 1; off < 16; off <<= 1)
        mx = fmaxf(mx, __shfl_xor(mx, off, 32));
      const float mn = fmaxf(rmax[r], mx);
      corr[r] = __expf(rmax[r] - mn);
      rmax[r] = mn;
      c0[r] = __expf(c0[r] - mn);
      c1[r] = __expf(c1[r] - mn);
      float sm = c0[r] + c1[r];
#pragma unroll
      for (int off = 1; off < 16; off <<= 1)
        sm += __shfl_xor(sm, off, 32);
      rsum[r] = rsum[r] * corr[r] + sm;
    }
#pragma unroll
    for (int nt = 0; nt < 4; ++nt)
#pragma unroll
      for (int r = 0; r < 8; ++r) oacc[nt][r] *= corr[r];

    // ---- stage P (C-layout f32) -> LDS bf16 for A-layout reload ----
    __bf16* pw = ldsP[wave];
#pragma unroll
    for (int r = 0; r < 8; ++r) {
      const int row = hf * 8 + r;
      pw[row * 32 + l16]      = (__bf16)c0[r];
      pw[row * 32 + 16 + l16] = (__bf16)c1[r];
    }
    __syncthreads();   // V tile + own P visible

    // ---- P A-fragment (16x32) ----
    v16bf pa = load_afrag(pw + (size_t)l16 * 32 + hf * 16);

    // ---- O(16x64) += P(16x32) @ V(32x64): 4 WMMA n-tiles ----
#pragma unroll
    for (int nt = 0; nt < 4; ++nt) {
      const __bf16* vp = ldsV + (size_t)(nt * 16 + l16) * 32 + hf * 16;
      v8bf v0 = *(const v8bf*)(vp);
      v8bf v1 = *(const v8bf*)(vp + 8);
      v16bf vb;
#pragma unroll
      for (int e = 0; e < 8; ++e) { vb[e] = v0[e]; vb[e + 8] = v1[e]; }
      oacc[nt] = wmma_bf16(pa, vb, oacc[nt]);
    }
  }

  // ---- epilogue: normalize and write bf16 [b*T+t, h*HD + hd] ----
#pragma unroll
  for (int r = 0; r < 8; ++r) {
    const float inv = 1.0f / rsum[r];
    const int t = t0 + hf * 8 + r;
    __bf16* op = obuf + (size_t)(b * T + t) * Dm + h * HD;
#pragma unroll
    for (int nt = 0; nt < 4; ++nt)
      op[nt * 16 + l16] = (__bf16)(oacc[nt][r] * inv);
  }
}

// ---------------------------------------------------------------------------
// Host launcher
// ---------------------------------------------------------------------------
extern "C" void kernel_launch(void* const* d_in, const int* in_sizes, int n_in,
                              void* d_out, int out_size, void* d_ws, size_t ws_size,
                              hipStream_t stream) {
  (void)in_sizes; (void)n_in; (void)out_size; (void)ws_size;

  const int Bn = 4, T = 2048, S = 512, D = 1024, H = 16;

  const float* x    = (const float*)d_in[0];
  const float* ctx  = (const float*)d_in[1];
  const unsigned char* cm = (const unsigned char*)d_in[2];  // bool mask, 1B each
  const float* Wq   = (const float*)d_in[3];
  const float* bq   = (const float*)d_in[4];
  const float* Wkv  = (const float*)d_in[5];
  const float* bkv  = (const float*)d_in[6];
  const float* Wp   = (const float*)d_in[7];
  const float* bp   = (const float*)d_in[8];
  float* out = (float*)d_out;

  // workspace (bf16 elements)
  __bf16* qbuf  = (__bf16*)d_ws;                        // [B*T, D]     8M
  __bf16* kvbuf = qbuf  + (size_t)Bn * T * D;           // [B*S, 2D]    4M
  __bf16* abuf  = kvbuf + (size_t)Bn * S * 2 * D;       // [B*T, D]     8M
  __bf16* xbf   = abuf  + (size_t)Bn * T * D;           // [B*T, D]     8M
  __bf16* cbf   = xbf   + (size_t)Bn * T * D;           // [B*S, D]     2M
  __bf16* WqT   = cbf   + (size_t)Bn * S * D;           // [D, D]       1M
  __bf16* WkvT  = WqT   + (size_t)D * D;                // [2D, D]      2M
  __bf16* WpT   = WkvT  + (size_t)2 * D * D;            // [D, D]       1M

  dim3 blk(256);
  dim3 tblk(32, 8);

  // activations -> bf16
  {
    size_t nx = (size_t)Bn * T * D, nc = (size_t)Bn * S * D;
    cvt_f32_bf16<<<dim3((unsigned)(nx / 8 / 256)), blk, 0, stream>>>(x, xbf, nx);
    cvt_f32_bf16<<<dim3((unsigned)(nc / 8 / 256)), blk, 0, stream>>>(ctx, cbf, nc);
  }
  // weights -> transposed bf16 [N,K]
  transpose_f32_bf16<<<dim3(D / 32, D / 32), tblk, 0, stream>>>(Wq, WqT, D, D);
  transpose_f32_bf16<<<dim3(2 * D / 32, D / 32), tblk, 0, stream>>>(Wkv, WkvT, D, 2 * D);
  transpose_f32_bf16<<<dim3(D / 32, D / 32), tblk, 0, stream>>>(Wp, WpT, D, D);

  // Q = x @ Wq + bq   -> bf16
  gemm_bf16_t<__bf16><<<dim3((Bn * T) / 64, D / 128), blk, 0, stream>>>(
      xbf, WqT, bq, qbuf, Bn * T, D, D);

  // KV = context @ Wkv + bkv -> bf16
  gemm_bf16_t<__bf16><<<dim3((Bn * S) / 64, (2 * D) / 128), blk, 0, stream>>>(
      cbf, WkvT, bkv, kvbuf, Bn * S, 2 * D, D);

  // flash attention -> bf16 [B,T,H*Hd]
  attn_kernel<<<dim3(T / 128, Bn * H), blk, 0, stream>>>(
      qbuf, kvbuf, cm, abuf, Bn, T, S, H);

  // out = attn_out @ Wp + bp -> fp32
  gemm_bf16_t<float><<<dim3((Bn * T) / 64, D / 128), blk, 0, stream>>>(
      abuf, WpT, bp, out, Bn * T, D, D);
}
